// Model_17789754540572
// MI455X (gfx1250) — matrix operand order repaired and checked by reading, not compiled, not run on hardware
//
#include <hip/hip_runtime.h>

// ---------------------------------------------------------------------------
// Flash-attention (fused QK^T -> online softmax -> dropout -> PV) for
// MI455X / gfx1250, wave32, bf16 WMMA with f32 accumulation.
// TDM (tensor_load_to_lds) pipelines K/V tile staging against WMMA compute.
// B=32, Sq=Skv=2048, D=128.
// ---------------------------------------------------------------------------

typedef __bf16 bf16_t;
typedef __attribute__((ext_vector_type(16))) __bf16 v16bf;
typedef __attribute__((ext_vector_type(8)))  __bf16 v8bf;
typedef __attribute__((ext_vector_type(8)))  float  v8f;
typedef __attribute__((ext_vector_type(4)))  unsigned int u32x4;
typedef __attribute__((ext_vector_type(8)))  unsigned int u32x8;

#define B_BATCH   32
#define SQ        2048
#define SKV       2048
#define DH        128
#define BLOCK_M   128     // query rows per workgroup (8 waves x 16 rows)
#define KV_T      64      // kv columns per inner tile
#define NT        (SKV / KV_T)
#define NWAVES    8
#define VT_STRIDE 72      // padded LDS stride for transposed V (16B aligned rows)
#define KEEP_P    0.7f

// Issue a 2-D TDM tile load: rows x cols f32 elements, row stride `stride`
// elements, from global `gsrc` into LDS byte offset `lds_off`.
// Descriptor layout per CDNA5 ISA section 8.3/8.4 (D# groups 0 and 1).
__device__ __forceinline__ void tdm_load_2d(const void* gsrc, unsigned lds_off,
                                            unsigned rows, unsigned cols,
                                            unsigned stride_elems)
{
    const unsigned long long ga = (unsigned long long)(size_t)gsrc;

    u32x4 g0;
    g0[0] = 1u;                                    // count=1 (valid user D#)
    g0[1] = lds_off;                               // lds_addr (bytes)
    g0[2] = (unsigned)(ga & 0xFFFFFFFFu);          // global_addr[31:0]
    g0[3] = (unsigned)((ga >> 32) & 0x1FFFFFFu)    // global_addr[56:32]
          | (2u << 30);                            // type = 2 ("image")

    u32x8 g1;
    g1[0] = (2u << 16);                            // wg_mask=0, data_size=4B (code 2)
    g1[1] = (cols & 0xFFFFu) << 16;                // tensor_dim0[15:0]
    g1[2] = ((cols >> 16) & 0xFFFFu)               // tensor_dim0[31:16]
          | ((rows & 0xFFFFu) << 16);              // tensor_dim1[15:0]
    g1[3] = ((rows >> 16) & 0xFFFFu)               // tensor_dim1[31:16]
          | ((cols & 0xFFFFu) << 16);              // tile_dim0 = cols
    g1[4] = rows & 0xFFFFu;                        // tile_dim1 = rows, tile_dim2=0
    g1[5] = stride_elems;                          // tensor_dim0_stride[31:0]
    g1[6] = 0;                                     // stride hi / dim1_stride lo
    g1[7] = 0;

    asm volatile("tensor_load_to_lds %0, %1" :: "s"(g0), "s"(g1) : "memory");
}

__global__ __launch_bounds__(256)
void fattn_bf16_wmma(const float* __restrict__ Q,
                     const float* __restrict__ K,
                     const float* __restrict__ V,
                     const float* __restrict__ scale,
                     float* __restrict__ O)
{
    __shared__ __align__(16) float rawK[KV_T * DH];      // 32KB f32 TDM landing pad
    __shared__ __align__(16) float rawV[KV_T * DH];      // 32KB f32 TDM landing pad
    __shared__ bf16_t sK [KV_T * DH];                    // [kv][d]  bf16 row-major
    __shared__ bf16_t sVt[DH * VT_STRIDE];               // [d][kv]  bf16 transposed
    __shared__ bf16_t sP [NWAVES * 16 * KV_T];           // per-wave P tile [m][kv]

    const int tid  = threadIdx.x;
    const int wave = tid >> 5;
    const int lane = tid & 31;
    const int half = lane >> 4;      // 0: lanes 0-15, 1: lanes 16-31
    const int l16  = lane & 15;

    // Wave-uniform (SGPR) predicate: forces a scalar branch so waves 1..7
    // really jump over the TDM issue (TDM ignores EXEC, so exec-masking would
    // duplicate the DMA 8x).
    const bool tdm_wave = (__builtin_amdgcn_readfirstlane(tid) < 32);

    const int b     = blockIdx.y;
    const int qbase = blockIdx.x * BLOCK_M + wave * 16;

    const float sc = scale[b];

    const float* __restrict__ Qb = Q + (size_t)b * SQ  * DH;
    const float* __restrict__ Kb = K + (size_t)b * SKV * DH;
    const float* __restrict__ Vb = V + (size_t)b * SKV * DH;

    const unsigned ldsK = (unsigned)(size_t)&rawK[0];
    const unsigned ldsV = (unsigned)(size_t)&rawV[0];

    // ---- Q tile -> 4 bf16 A-fragments (16x32 each, covering D=128) --------
    v16bf aQ[4];
    {
        const float* qp = Qb + (size_t)(qbase + l16) * DH;
        #pragma unroll
        for (int kc = 0; kc < 4; ++kc) {
            #pragma unroll
            for (int v = 0; v < 8; ++v) {
                const int d = kc * 32 + half * 16 + 2 * v;
                const float2 f = *(const float2*)(qp + d);
                aQ[kc][2 * v]     = (__bf16)f.x;
                aQ[kc][2 * v + 1] = (__bf16)f.y;
            }
        }
    }

    // ---- accumulators + online-softmax state ------------------------------
    v8f o[8] = {};
    float m_run[8], l_run[8];
    #pragma unroll
    for (int r = 0; r < 8; ++r) { m_run[r] = -1e30f; l_run[r] = 0.0f; }

    const unsigned keep_thresh = (unsigned)(KEEP_P * 16777216.0f);

    // ---- prologue: TDM-load tile 0 (K and V) ------------------------------
    if (tdm_wave) {
        tdm_load_2d(Kb, ldsK, KV_T, DH, DH);
        tdm_load_2d(Vb, ldsV, KV_T, DH, DH);
    }

    for (int it = 0; it < NT; ++it) {
        const int kv0 = it * KV_T;

        if (tdm_wave) __builtin_amdgcn_s_wait_tensorcnt(0);
        __syncthreads();                       // raw tile `it` visible to all

        // ---- LDS->LDS convert: rawK -> sK (bf16), rawV -> sVt (transposed)
        #pragma unroll
        for (int i = 0; i < 8; ++i) {
            const int idx = (i * 256 + tid) * 4;
            const int row = idx >> 7;          // kv within tile
            const int col = idx & 127;         // d
            const float4 f = *(const float4*)&rawK[idx];
            bf16_t* dk = &sK[row * DH + col];
            dk[0] = (__bf16)f.x; dk[1] = (__bf16)f.y;
            dk[2] = (__bf16)f.z; dk[3] = (__bf16)f.w;
            const float4 g = *(const float4*)&rawV[idx];
            sVt[(col + 0) * VT_STRIDE + row] = (__bf16)g.x;
            sVt[(col + 1) * VT_STRIDE + row] = (__bf16)g.y;
            sVt[(col + 2) * VT_STRIDE + row] = (__bf16)g.z;
            sVt[(col + 3) * VT_STRIDE + row] = (__bf16)g.w;
        }
        __syncthreads();                       // bf16 tiles ready, raw pads free

        // ---- kick TDM for tile it+1 (overlaps with WMMA compute below) ----
        if (it + 1 < NT && tdm_wave) {
            const size_t nxt = (size_t)(kv0 + KV_T) * DH;
            tdm_load_2d(Kb + nxt, ldsK, KV_T, DH, DH);
            tdm_load_2d(Vb + nxt, ldsV, KV_T, DH, DH);
        }

        // ---- S = (Q @ K^T) * scale : 4 N-tiles of 16 kv columns -----------
        // Load all 4 B-fragments of a tile into distinct registers first so
        // the 8 ds_load_b128 are clustered behind one s_wait_dscnt, then the
        // 4 WMMAs issue back-to-back.
        v8f s[4];
        #pragma unroll
        for (int t = 0; t < 4; ++t) {
            v16bf bk[4];
            #pragma unroll
            for (int kc = 0; kc < 4; ++kc)
                bk[kc] = *(const v16bf*)
                         &sK[(t * 16 + l16) * DH + kc * 32 + half * 16];
            v8f acc = {};
            #pragma unroll
            for (int kc = 0; kc < 4; ++kc)
                acc = __builtin_amdgcn_wmma_f32_16x16x32_bf16(
                          false, aQ[kc], false, bk[kc], (short)0, acc,
                          false, false);
            s[t] = acc * sc;
        }

        // ---- online softmax over the 64 new columns -----------------------
        float alpha[8];
        #pragma unroll
        for (int r = 0; r < 8; ++r) {
            float mx = fmaxf(fmaxf(s[0][r], s[1][r]), fmaxf(s[2][r], s[3][r]));
            #pragma unroll
            for (int off = 8; off >= 1; off >>= 1)
                mx = fmaxf(mx, __shfl_xor(mx, off, 32));
            const float mnew = fmaxf(m_run[r], mx);
            alpha[r] = __expf(m_run[r] - mnew);
            m_run[r] = mnew;
            float psum = 0.0f;
            #pragma unroll
            for (int t = 0; t < 4; ++t) {
                const float p = __expf(s[t][r] - mnew);
                s[t][r] = p;
                psum += p;
            }
            #pragma unroll
            for (int off = 8; off >= 1; off >>= 1)
                psum += __shfl_xor(psum, off, 32);
            l_run[r] = l_run[r] * alpha[r] + psum;   // un-dropped normalizer
            #pragma unroll
            for (int t2 = 0; t2 < 8; ++t2) o[t2][r] *= alpha[r];
        }

        // ---- dropout + C-layout -> A-layout via per-wave LDS tile ---------
        bf16_t* pP = &sP[wave * 16 * KV_T];
        #pragma unroll
        for (int t = 0; t < 4; ++t) {
            const int kcol = kv0 + t * 16 + l16;
            #pragma unroll
            for (int r = 0; r < 8; ++r) {
                const int m    = half * 8 + r;
                const int qrow = qbase + m;
                unsigned x = (unsigned)(qrow * SKV + kcol) ^ ((unsigned)b * 0x9E3779B9u);
                x ^= x >> 16; x *= 0x7FEB352Du;
                x ^= x >> 15; x *= 0x846CA68Bu;
                x ^= x >> 16;
                float p = s[t][r];
                p = ((x & 0xFFFFFFu) < keep_thresh) ? p * (1.0f / KEEP_P) : 0.0f;
                pP[m * KV_T + t * 16 + l16] = (__bf16)p;
            }
        }

        // ---- O += P @ V (K-dim 64 = 2 chunks of 32) -----------------------
        #pragma unroll
        for (int c = 0; c < 2; ++c) {
            const bf16_t* p0 = &pP[l16 * KV_T + c * 32 + half * 16];
            const v8bf plo = *(const v8bf*)p0;
            const v8bf phi = *(const v8bf*)(p0 + 8);
            const v16bf aP = __builtin_shufflevector(plo, phi,
                                0,1,2,3,4,5,6,7,8,9,10,11,12,13,14,15);
            #pragma unroll
            for (int g = 0; g < 2; ++g) {
                v16bf bv[4];
                #pragma unroll
                for (int j = 0; j < 4; ++j) {
                    const int t2 = g * 4 + j;
                    const bf16_t* vp =
                        &sVt[(t2 * 16 + l16) * VT_STRIDE + c * 32 + half * 16];
                    const v8bf vlo = *(const v8bf*)vp;
                    const v8bf vhi = *(const v8bf*)(vp + 8);
                    bv[j] = __builtin_shufflevector(vlo, vhi,
                               0,1,2,3,4,5,6,7,8,9,10,11,12,13,14,15);
                }
                #pragma unroll
                for (int j = 0; j < 4; ++j)
                    o[g * 4 + j] = __builtin_amdgcn_wmma_f32_16x16x32_bf16(
                                       false, aP, false, bv[j], (short)0,
                                       o[g * 4 + j], false, false);
            }
        }
    }

    // ---- epilogue: O /= l, store fp32 --------------------------------------
    #pragma unroll
    for (int r = 0; r < 8; ++r) {
        const int qrow = qbase + half * 8 + r;
        const float inv = 1.0f / l_run[r];
        float* op = O + ((size_t)b * SQ + qrow) * DH;
        #pragma unroll
        for (int t2 = 0; t2 < 8; ++t2)
            op[t2 * 16 + l16] = o[t2][r] * inv;
    }
}

extern "C" void kernel_launch(void* const* d_in, const int* in_sizes, int n_in,
                              void* d_out, int out_size, void* d_ws, size_t ws_size,
                              hipStream_t stream) {
    (void)in_sizes; (void)n_in; (void)out_size; (void)d_ws; (void)ws_size;
    const float* q  = (const float*)d_in[0];
    const float* k  = (const float*)d_in[1];
    const float* v  = (const float*)d_in[2];
    const float* sf = (const float*)d_in[3];
    float* out = (float*)d_out;

    dim3 grid(SQ / BLOCK_M, B_BATCH);   // (16, 32)
    fattn_bf16_wmma<<<grid, 256, 0, stream>>>(q, k, v, sf, out);
}
